// RelPositionMultiHeadAttention_83863531422238
// MI455X (gfx1250) — matrix operand order repaired and checked by reading, not compiled, not run on hardware
//
#include <hip/hip_runtime.h>
#include <hip/hip_bf16.h>
#include <math.h>

#define BDIM 4
#define TDIM 2048
#define DMODEL 512
#define NHEAD 8
#define DHEAD 64
#define MROWS (BDIM * TDIM) /* 8192 */

typedef __attribute__((ext_vector_type(16))) __bf16 bf16x16;
typedef __attribute__((ext_vector_type(8)))  __bf16 bf16x8;
typedef __attribute__((ext_vector_type(8)))  float  f32x8;

static __device__ __forceinline__ f32x8 wmma_bf16(bf16x16 a, bf16x16 b, f32x8 c) {
  // D = A(16x32) * B(32x16) + C, f32 accumulate
  return __builtin_amdgcn_wmma_f32_16x16x32_bf16(false, a, false, b, (short)0, c,
                                                 false, false);
}

static __device__ __forceinline__ bf16x16 cat8(bf16x8 lo, bf16x8 hi) {
  bf16x16 r;
#pragma unroll
  for (int i = 0; i < 8; ++i) { r[i] = lo[i]; r[8 + i] = hi[i]; }
  return r;
}

// ---------------------------------------------------------------------------
// Pass 0a: transpose + convert the five 512x512 f32 weights to bf16 [N][K]
// so WMMA B-fragments are contiguous 32B loads per lane.
// ---------------------------------------------------------------------------
__global__ void wt_kernel(const float* __restrict__ Wq, const float* __restrict__ Wk,
                          const float* __restrict__ Wv, const float* __restrict__ Wp,
                          const float* __restrict__ Wo, __bf16* __restrict__ Wt) {
  int idx = blockIdx.x * 256 + threadIdx.x;       // 5 * 512 * 512 total
  int w = idx >> 18;
  int e = idx & 262143;
  int n = e >> 9;
  int k = e & 511;
  const float* src = (w == 0) ? Wq : (w == 1) ? Wk : (w == 2) ? Wv : (w == 3) ? Wp : Wo;
  Wt[(size_t)w * 262144 + (size_t)n * DMODEL + k] = (__bf16)src[(size_t)k * DMODEL + n];
}

// ---------------------------------------------------------------------------
// Pass 0b: one-time f32 -> bf16 conversion of x and pos (read once, used 4x)
// ---------------------------------------------------------------------------
__global__ void cvt_kernel(const float* __restrict__ X, const float* __restrict__ P,
                           __bf16* __restrict__ Xb, __bf16* __restrict__ Pbm) {
  const size_t NV = (size_t)MROWS * DMODEL / 8;   // vec8 count per tensor
  size_t idx = (size_t)blockIdx.x * 256 + threadIdx.x;  // 2*NV threads
  const float* src = (idx < NV) ? X : P;
  __bf16* dst = (idx < NV) ? Xb : Pbm;
  const size_t v = (idx < NV) ? idx : idx - NV;
  const f32x8 f = *(const f32x8*)(src + v * 8);
  bf16x8 o;
#pragma unroll
  for (int i = 0; i < 8; ++i) o[i] = (__bf16)f[i];
  *(bf16x8*)(dst + v * 8) = o;
}

// ---------------------------------------------------------------------------
// Pass 1: projection GEMM  Y[8192,512] = Xb @ W + b, written head-major bf16.
// mode 0: Q  -> out0 = Qu (+pos_bias_u), out1 = Qv (+pos_bias_v), [B,H,T,64]
// mode 1: K  -> out0 [B,H,T,64]
// mode 2: V  -> out0 TRANSPOSED [B,H,64,T]
// mode 3: P  -> out0 [B,H,T,64] (no bias)
// 128 thr / 4 waves; each wave computes a 16x64 strip (4 C-frags sharing one
// A-frag per K-step -> 4 wmma per 6 fragment loads). Block tile = 16M x 256N.
// ---------------------------------------------------------------------------
__global__ void proj_kernel(const __bf16* __restrict__ X,
                            const __bf16* __restrict__ Wt,
                            const float* __restrict__ bias,
                            const float* __restrict__ pbu,
                            const float* __restrict__ pbv,
                            __bf16* __restrict__ out0,
                            __bf16* __restrict__ out1,
                            int mode) {
  const int lane = threadIdx.x & 31;
  const int wave = threadIdx.x >> 5;
  const int l16  = lane & 15;
  const int half = lane >> 4;
  const int m0 = (blockIdx.x >> 1) * 16;
  const int nb = (blockIdx.x & 1) * 256 + wave * 64;
  const __bf16* arow = X + (size_t)(m0 + l16) * DMODEL;

  f32x8 acc[4];
#pragma unroll
  for (int j = 0; j < 4; ++j) acc[j] = (f32x8){0.f,0.f,0.f,0.f,0.f,0.f,0.f,0.f};

#pragma unroll 2
  for (int kk = 0; kk < DMODEL; kk += 32) {
    // A-fragment: lane<16 -> K {0..15}; lane>=16 -> K {16..31}
    const bf16x16 a = cat8(*(const bf16x8*)(arow + kk + half * 16),
                           *(const bf16x8*)(arow + kk + 8 + half * 16));
#pragma unroll
    for (int j = 0; j < 4; ++j) {
      const __bf16* wrow = Wt + (size_t)(nb + j * 16 + l16) * DMODEL;
      const bf16x16 b = *(const bf16x16*)(wrow + kk + half * 16);
      acc[j] = wmma_bf16(a, b, acc[j]);
    }
  }

#pragma unroll
  for (int j = 0; j < 4; ++j) {
    const int ncol = nb + j * 16 + l16;
    const int h  = ncol >> 6;
    const int dh = ncol & 63;
    const float bval = bias ? bias[ncol] : 0.f;
#pragma unroll
    for (int r = 0; r < 8; ++r) {
      const int m = m0 + r + half * 8;
      const int bidx = m / TDIM;
      const int t = m % TDIM;
      const float v = acc[j][r] + bval;
      const size_t bh = (size_t)bidx * NHEAD + h;
      if (mode == 0) {
        out0[(bh * TDIM + t) * DHEAD + dh] = (__bf16)(v + pbu[h * DHEAD + dh]);
        out1[(bh * TDIM + t) * DHEAD + dh] = (__bf16)(v + pbv[h * DHEAD + dh]);
      } else if (mode == 2) {
        out0[(bh * DHEAD + dh) * TDIM + t] = (__bf16)v;   // V stored transposed
      } else {
        out0[(bh * TDIM + t) * DHEAD + dh] = (__bf16)v;
      }
    }
  }
}

// ---------------------------------------------------------------------------
// Pass 2: flash attention with exact relative-shift.
// One workgroup (128 thr / 4 waves) per (b, h, 16-row t-tile).
// LDS: PS band [17][T] bf16 (69632B) + P stage (4096B) + m/l (512B) + ctx (16384B)
// ---------------------------------------------------------------------------
__global__ void __launch_bounds__(128)
attn_kernel(const __bf16* __restrict__ Qu, const __bf16* __restrict__ Qv,
            const __bf16* __restrict__ Kb, const __bf16* __restrict__ Vt,
            const __bf16* __restrict__ Pb, const unsigned char* __restrict__ mask,
            __bf16* __restrict__ ctx_out) {
  extern __shared__ char smem[];
  __bf16* PS    = (__bf16*)smem;              // 17 * 2048 bf16
  __bf16* stage = PS + 17 * TDIM;             // 4 * 16 * 32 bf16
  float*  red_m = (float*)(stage + 4 * 512);  // 4 * 16
  float*  red_l = red_m + 64;                 // 4 * 16
  float*  cbuf  = red_l + 64;                 // 4 * 16 * 64

  const int lane = threadIdx.x & 31;
  const int wave = threadIdx.x >> 5;
  const int l16  = lane & 15;
  const int half = lane >> 4;
  const int bh = blockIdx.x >> 7;             // 128 t-tiles per (b,h)
  const int tt = blockIdx.x & 127;
  const int t0 = tt * 16;
  const int b  = bh >> 3;
  const int h  = bh & 7;

  const __bf16* Qub = Qu + (size_t)bh * TDIM * DHEAD;
  const __bf16* Qvb = Qv + (size_t)bh * TDIM * DHEAD;
  const __bf16* Kbb = Kb + (size_t)bh * TDIM * DHEAD;
  const __bf16* Pbb = Pb + (size_t)bh * TDIM * DHEAD;
  const __bf16* Vtb = Vt + (size_t)bh * DHEAD * TDIM;

  // ---- Phase A: PS[t, j] = Qv[t] . P[j] for t in [t0, t0+16], all j --------
  const int tr = t0 + l16;
  const bf16x16 qv0 = cat8(*(const bf16x8*)(Qvb + (size_t)tr * DHEAD + half * 16),
                           *(const bf16x8*)(Qvb + (size_t)tr * DHEAD + 8 + half * 16));
  const bf16x16 qv1 = cat8(*(const bf16x8*)(Qvb + (size_t)tr * DHEAD + 32 + half * 16),
                           *(const bf16x8*)(Qvb + (size_t)tr * DHEAD + 40 + half * 16));
  for (int j0 = wave * 16; j0 < TDIM; j0 += 64) {
    const int j = j0 + l16;
    const bf16x16 pb0 = *(const bf16x16*)(Pbb + (size_t)j * DHEAD + half * 16);
    const bf16x16 pb1 = *(const bf16x16*)(Pbb + (size_t)j * DHEAD + 32 + half * 16);
    f32x8 acc = {0.f, 0.f, 0.f, 0.f, 0.f, 0.f, 0.f, 0.f};
    acc = wmma_bf16(qv0, pb0, acc);
    acc = wmma_bf16(qv1, pb1, acc);
#pragma unroll
    for (int r = 0; r < 8; ++r)
      PS[(size_t)(r + half * 8) * TDIM + j0 + l16] = (__bf16)acc[r];
  }
  // extra band row t0+16 (needed by row t0+15's wrapped shift), cheap VALU dot
  if (t0 + 16 < TDIM) {
    for (int j = threadIdx.x; j < TDIM; j += 128) {
      float s = 0.f;
#pragma unroll 8
      for (int d = 0; d < DHEAD; ++d)
        s += (float)Qvb[(size_t)(t0 + 16) * DHEAD + d] * (float)Pbb[(size_t)j * DHEAD + d];
      PS[(size_t)16 * TDIM + j] = (__bf16)s;
    }
  }
  __syncthreads();

  // ---- Phase B: flash loop over s --------------------------------------
  const bf16x16 qu0 = cat8(*(const bf16x8*)(Qub + (size_t)tr * DHEAD + half * 16),
                           *(const bf16x8*)(Qub + (size_t)tr * DHEAD + 8 + half * 16));
  const bf16x16 qu1 = cat8(*(const bf16x8*)(Qub + (size_t)tr * DHEAD + 32 + half * 16),
                           *(const bf16x8*)(Qub + (size_t)tr * DHEAD + 40 + half * 16));

  f32x8 ctx[4];
#pragma unroll
  for (int dt = 0; dt < 4; ++dt) ctx[dt] = (f32x8){0.f,0.f,0.f,0.f,0.f,0.f,0.f,0.f};
  float mrow[8], lrow[8];
#pragma unroll
  for (int r = 0; r < 8; ++r) { mrow[r] = -3.0e38f; lrow[r] = 0.f; }

  __bf16* st = stage + wave * 512; // per-wave 16x32 bf16 staging

  for (int s0 = wave * 32; s0 < TDIM; s0 += 128) {
    if (s0 + 128 < TDIM) {
      __builtin_prefetch(Kbb + (size_t)(s0 + 128 + l16) * DHEAD, 0, 1);
      __builtin_prefetch(Vtb + (size_t)l16 * TDIM + s0 + 128, 0, 1);
    }
    // content scores: two 16x16 column sub-tiles
    f32x8 sc[2];
#pragma unroll
    for (int sub = 0; sub < 2; ++sub) {
      const int ss = s0 + sub * 16 + l16;
      const bf16x16 kb0 = *(const bf16x16*)(Kbb + (size_t)ss * DHEAD + half * 16);
      const bf16x16 kb1 = *(const bf16x16*)(Kbb + (size_t)ss * DHEAD + 32 + half * 16);
      f32x8 a = {0.f, 0.f, 0.f, 0.f, 0.f, 0.f, 0.f, 0.f};
      a = wmma_bf16(qu0, kb0, a);
      a = wmma_bf16(qu1, kb1, a);
      sc[sub] = a;
    }
    // add rel-position (shifted), mask, scale
    float sv[2][8];
#pragma unroll
    for (int sub = 0; sub < 2; ++sub) {
#pragma unroll
      for (int r = 0; r < 8; ++r) {
        const int t = t0 + r + half * 8;
        const int s = s0 + sub * 16 + l16;
        float pos;
        if (s <= t)          pos = (float)PS[(size_t)(t - t0) * TDIM + (TDIM - 1 + s - t)];
        else if (s == t + 1) pos = 0.f;
        else                 pos = (float)PS[(size_t)(t + 1 - t0) * TDIM + (s - t - 2)];
        float v = (sc[sub][r] + pos) * 0.125f;
        if (!mask[((size_t)b * TDIM + t) * TDIM + s]) v = -3.0e38f;
        sv[sub][r] = v;
      }
    }
    // online softmax per row (row lives in a 16-lane group -> shfl_xor reduce)
    float pexp[2][8];
#pragma unroll
    for (int r = 0; r < 8; ++r) {
      float rm = fmaxf(sv[0][r], sv[1][r]);
      rm = fmaxf(rm, __shfl_xor(rm, 1));
      rm = fmaxf(rm, __shfl_xor(rm, 2));
      rm = fmaxf(rm, __shfl_xor(rm, 4));
      rm = fmaxf(rm, __shfl_xor(rm, 8));
      const float mnew = fmaxf(mrow[r], rm);
      const float corr = __expf(mrow[r] - mnew);
      float rs = 0.f;
#pragma unroll
      for (int sub = 0; sub < 2; ++sub) {
        const float e = __expf(sv[sub][r] - mnew);
        pexp[sub][r] = e;
        rs += e;
      }
      rs += __shfl_xor(rs, 1);
      rs += __shfl_xor(rs, 2);
      rs += __shfl_xor(rs, 4);
      rs += __shfl_xor(rs, 8);
      lrow[r] = lrow[r] * corr + rs;
      mrow[r] = mnew;
#pragma unroll
      for (int dt = 0; dt < 4; ++dt) ctx[dt][r] *= corr;
    }
    // stage P (C-layout) to LDS, re-read as A-fragment (same wave: in-order DS)
#pragma unroll
    for (int sub = 0; sub < 2; ++sub)
#pragma unroll
      for (int r = 0; r < 8; ++r)
        st[(r + half * 8) * 32 + sub * 16 + l16] = (__bf16)pexp[sub][r];
    const bf16x16 pa = cat8(*(const bf16x8*)(st + l16 * 32 + half * 16),
                            *(const bf16x8*)(st + l16 * 32 + 8 + half * 16));
    // ctx += P @ V  (V transposed in memory: contiguous along s)
#pragma unroll
    for (int dt = 0; dt < 4; ++dt) {
      const int d = dt * 16 + l16;
      const bf16x16 vb = *(const bf16x16*)(Vtb + (size_t)d * TDIM + s0 + half * 16);
      ctx[dt] = wmma_bf16(pa, vb, ctx[dt]);
    }
  }

  // ---- cross-wave merge -------------------------------------------------
  if (l16 == 0) {
#pragma unroll
    for (int r = 0; r < 8; ++r) {
      red_m[wave * 16 + half * 8 + r] = mrow[r];
      red_l[wave * 16 + half * 8 + r] = lrow[r];
    }
  }
#pragma unroll
  for (int dt = 0; dt < 4; ++dt)
#pragma unroll
    for (int r = 0; r < 8; ++r)
      cbuf[wave * 1024 + (r + half * 8) * 64 + dt * 16 + l16] = ctx[dt][r];
  __syncthreads();

  for (int idx = threadIdx.x; idx < 1024; idx += 128) {
    const int row = idx >> 6;
    const int d   = idx & 63;
    float M = -3.0e38f;
#pragma unroll
    for (int w = 0; w < 4; ++w) M = fmaxf(M, red_m[w * 16 + row]);
    float L = 0.f, C = 0.f;
#pragma unroll
    for (int w = 0; w < 4; ++w) {
      const float f = __expf(red_m[w * 16 + row] - M);
      L += red_l[w * 16 + row] * f;
      C += cbuf[w * 1024 + row * 64 + d] * f;
    }
    ctx_out[((size_t)b * TDIM + t0 + row) * DMODEL + h * DHEAD + d] = (__bf16)(C / L);
  }
}

// ---------------------------------------------------------------------------
// Pass 3: out = ctx @ Wo + bo  (bf16 WMMA, f32 output), 16x64 strip per wave
// ---------------------------------------------------------------------------
__global__ void out_gemm(const __bf16* __restrict__ A, const __bf16* __restrict__ Wt,
                         const float* __restrict__ bo, float* __restrict__ out) {
  const int lane = threadIdx.x & 31;
  const int wave = threadIdx.x >> 5;
  const int l16  = lane & 15;
  const int half = lane >> 4;
  const int m0 = (blockIdx.x >> 1) * 16;
  const int nb = (blockIdx.x & 1) * 256 + wave * 64;
  const __bf16* arow = A + (size_t)(m0 + l16) * DMODEL;

  f32x8 acc[4];
#pragma unroll
  for (int j = 0; j < 4; ++j) acc[j] = (f32x8){0.f,0.f,0.f,0.f,0.f,0.f,0.f,0.f};

#pragma unroll 2
  for (int kk = 0; kk < DMODEL; kk += 32) {
    const bf16x16 a = cat8(*(const bf16x8*)(arow + kk + half * 16),
                           *(const bf16x8*)(arow + kk + 8 + half * 16));
#pragma unroll
    for (int j = 0; j < 4; ++j) {
      const __bf16* wrow = Wt + (size_t)(nb + j * 16 + l16) * DMODEL;
      const bf16x16 b = *(const bf16x16*)(wrow + kk + half * 16);
      acc[j] = wmma_bf16(a, b, acc[j]);
    }
  }

#pragma unroll
  for (int j = 0; j < 4; ++j) {
    const int ncol = nb + j * 16 + l16;
    const float bval = bo[ncol];
#pragma unroll
    for (int r = 0; r < 8; ++r)
      out[(size_t)(m0 + r + half * 8) * DMODEL + ncol] = acc[j][r] + bval;
  }
}

// ---------------------------------------------------------------------------
extern "C" void kernel_launch(void* const* d_in, const int* in_sizes, int n_in,
                              void* d_out, int out_size, void* d_ws, size_t ws_size,
                              hipStream_t stream) {
  const float* x    = (const float*)d_in[0];
  const float* pos  = (const float*)d_in[1];
  const unsigned char* mask = (const unsigned char*)d_in[2];
  const float* Wq   = (const float*)d_in[3];
  const float* bq   = (const float*)d_in[4];
  const float* Wk   = (const float*)d_in[5];
  const float* bk   = (const float*)d_in[6];
  const float* Wv   = (const float*)d_in[7];
  const float* bv   = (const float*)d_in[8];
  const float* Wpos = (const float*)d_in[9];
  const float* Wo   = (const float*)d_in[10];
  const float* bo   = (const float*)d_in[11];
  const float* pbu  = (const float*)d_in[12];
  const float* pbv  = (const float*)d_in[13];

  const size_t WT_ELEMS  = (size_t)5 * DMODEL * DMODEL;         // 1,310,720 bf16
  const size_t TEN_ELEMS = (size_t)BDIM * NHEAD * TDIM * DHEAD; // 4,194,304 bf16

  char* wsp = (char*)d_ws;
  __bf16* Wt   = (__bf16*)wsp;  wsp += WT_ELEMS  * sizeof(__bf16);
  __bf16* Xb   = (__bf16*)wsp;  wsp += TEN_ELEMS * sizeof(__bf16);
  __bf16* Posb = (__bf16*)wsp;  wsp += TEN_ELEMS * sizeof(__bf16);
  __bf16* Qu   = (__bf16*)wsp;  wsp += TEN_ELEMS * sizeof(__bf16);
  __bf16* Qv   = (__bf16*)wsp;  wsp += TEN_ELEMS * sizeof(__bf16);
  __bf16* Kbf  = (__bf16*)wsp;  wsp += TEN_ELEMS * sizeof(__bf16);
  __bf16* Vt   = (__bf16*)wsp;  wsp += TEN_ELEMS * sizeof(__bf16);
  __bf16* Pbf  = (__bf16*)wsp;  wsp += TEN_ELEMS * sizeof(__bf16);
  __bf16* Ctx  = (__bf16*)wsp;  wsp += TEN_ELEMS * sizeof(__bf16);

  // Pass 0: weight transpose + bf16; activation f32 -> bf16
  wt_kernel<<<dim3(5120), dim3(256), 0, stream>>>(Wq, Wk, Wv, Wpos, Wo, Wt);
  cvt_kernel<<<dim3(4096), dim3(256), 0, stream>>>(x, pos, Xb, Posb);

  // Pass 1: projections (q produces both Qu and Qv with the two biases fused)
  const int GPROJ = (MROWS / 16) * (DMODEL / 256); // 1024
  proj_kernel<<<dim3(GPROJ), dim3(128), 0, stream>>>(Xb,   Wt + 0 * 262144, bq, pbu, pbv, Qu, Qv, 0);
  proj_kernel<<<dim3(GPROJ), dim3(128), 0, stream>>>(Xb,   Wt + 1 * 262144, bk, nullptr, nullptr, Kbf, nullptr, 1);
  proj_kernel<<<dim3(GPROJ), dim3(128), 0, stream>>>(Xb,   Wt + 2 * 262144, bv, nullptr, nullptr, Vt, nullptr, 2);
  proj_kernel<<<dim3(GPROJ), dim3(128), 0, stream>>>(Posb, Wt + 3 * 262144, nullptr, nullptr, nullptr, Pbf, nullptr, 3);

  // Pass 2: flash attention with rel-shift (LDS: 69632 + 4096 + 512 + 16384)
  const int GATT = BDIM * NHEAD * (TDIM / 16); // 4096
  const size_t SMEM = (size_t)17 * TDIM * 2 + 4 * 512 * 2 + 2 * 64 * 4 + 4 * 1024 * 4; // 90624
  attn_kernel<<<dim3(GATT), dim3(128), SMEM, stream>>>(Qu, Qv, Kbf, Vt, Pbf, mask, Ctx);

  // Pass 3: output projection
  out_gemm<<<dim3(GPROJ), dim3(128), 0, stream>>>(Ctx, Wt + 4 * 262144, bo, (float*)d_out);

  (void)in_sizes; (void)n_in; (void)out_size; (void)ws_size;
}
